// WindObserverModule_80839874445994
// MI455X (gfx1250) — matrix operand order repaired and checked by reading, not compiled, not run on hardware
//
#include <hip/hip_runtime.h>
#include <hip/hip_bf16.h>

typedef _Float16 v16h __attribute__((ext_vector_type(16)));
typedef float    v8f  __attribute__((ext_vector_type(8)));
typedef float    v4f  __attribute__((ext_vector_type(4)));
typedef float    v2f  __attribute__((ext_vector_type(2)));

#define HDIM 32
#define ROWS 20   // padded LDS row stride in dwords (80B: 16B-aligned, conflict-free)

template <bool B> struct bconst { static constexpr bool value = B; };

__device__ __forceinline__ float fast_tanh(float x) {
#if __has_builtin(__builtin_amdgcn_tanhf)
  return __builtin_amdgcn_tanhf(x);           // gfx1250 v_tanh_f32
#else
  float e = __expf(2.0f * x);                 // v_exp_f32 path
  return 1.0f - 2.0f / (e + 1.0f);
#endif
}

// Load one 16x32 f16 A-tile from a row-major [*,32] f32 weight matrix.
// A-layout (16-bit 16x32): lane L holds row M=L%16; element pair (2v,2v+1)
// holds K = ((v<4)?2v:2v+8) + 8*(L/16) + {0,1}.
__device__ __forceinline__ v16h load_a_tile(const float* W, int rowOff, int Mlane, int hiK) {
  v16h a;
#pragma unroll
  for (int i = 0; i < 16; ++i) {
    int v = i >> 1;
    int k = ((v < 4) ? 2 * v : 2 * v + 8) + 8 * hiK + (i & 1);
    a[i] = (_Float16)W[(rowOff + Mlane) * HDIM + k];
  }
  return a;
}

__global__ __launch_bounds__(32)
void rnn_scan_kernel(const float* __restrict__ obs,
                     const float* __restrict__ h_obs,
                     const float* __restrict__ masks,
                     const float* __restrict__ W_ih,
                     const float* __restrict__ W_hh,
                     const float* __restrict__ b_ih,
                     const float* __restrict__ b_hh,
                     const float* __restrict__ W_mu,
                     const float* __restrict__ b_mu,
                     const float* __restrict__ W_lv,
                     const float* __restrict__ b_lv,
                     float* __restrict__ out,
                     int N, int T) {
  __shared__ float hT[HDIM * ROWS];   // h^T staging: row j (hidden), col e (env)
  __shared__ float mlds[16];          // per-env masks for broadcast

  const int lane = threadIdx.x;       // wave32
  const int e    = lane & 15;         // env-in-tile (D-layout column / A row)
  const int hiK  = lane >> 4;
  const int base = blockIdx.x * 16;

  float* out_h = out + (size_t)T * N * 4;

  // ---- loop-invariant weight tiles (A operands, resident in VGPRs) ----
  v16h aih0 = load_a_tile(W_ih, 0,  e, hiK);
  v16h aih1 = load_a_tile(W_ih, 16, e, hiK);
  v16h ahh0 = load_a_tile(W_hh, 0,  e, hiK);
  v16h ahh1 = load_a_tile(W_hh, 16, e, hiK);
  // head tile rows: M=0->W_mu[0], M=1->W_mu[1], M=8->W_lv[0], M=9->W_lv[1], rest 0.
  // lv at rows 8,9 => D lands as: lanes 0-15 VGPR0/1 = (mu0,mu1),
  // lanes 16-31 VGPR0/1 = (lv0,lv1) -> one unpredicated b64 store per step.
  v16h ahd;
  {
    const float* hrow =
        (e == 0) ? W_mu : (e == 1) ? (W_mu + HDIM)
      : (e == 8) ? W_lv : (e == 9) ? (W_lv + HDIM) : (const float*)0;
#pragma unroll
    for (int i = 0; i < 16; ++i) {
      int v = i >> 1;
      int k = ((v < 4) ? 2 * v : 2 * v + 8) + 8 * hiK + (i & 1);
      ahd[i] = hrow ? (_Float16)hrow[k] : (_Float16)0.0f;
    }
  }

  // ---- bias C tiles (f32 C/D layout: VGPR r -> row j = tile*16 + r + 8*hiK) ----
  v8f cb0, cb1, chd;
#pragma unroll
  for (int r = 0; r < 8; ++r) {
    int j0 = r + 8 * hiK;
    cb0[r] = b_ih[j0] + b_hh[j0];
    cb1[r] = b_ih[16 + j0] + b_hh[16 + j0];
    float hv = 0.0f;
    if (r == 0) hv = hiK ? b_lv[0] : b_mu[0];
    if (r == 1) hv = hiK ? b_lv[1] : b_mu[1];
    chd[r] = hv;
  }

  // ---- initial hidden state in D layout (rows=hidden, cols=env) ----
  v8f h0, h1;
#pragma unroll
  for (int r = 0; r < 8; ++r) {
    h0[r] = h_obs[(size_t)(base + e) * HDIM + r + 8 * hiK];
    h1[r] = h_obs[(size_t)(base + e) * HDIM + 16 + r + 8 * hiK];
  }

  // ---- loop-carried pointers (uniform bumps; lane part stays in voffset) ----
  const float* xrow = obs + (size_t)base * HDIM;   // step-t obs block
  const float* mrow = masks + base;                // step-t masks block
  float*       orow = out;                         // head-output block for step t-1
  const size_t ostep = (size_t)N * 2;
  const size_t voff  = (hiK ? (size_t)T * N * 2 : 0) + (size_t)(base + e) * 2;

  // ---- software prefetch of x_0 / m_0 (x in B-layout: lane = env N, elements = feature K) ----
  float xc[16], xn[16], mc, mn;
#pragma unroll
  for (int q = 0; q < 16; ++q)
    xn[q] = __builtin_nontemporal_load(xrow + e * HDIM + ((q < 8) ? q : q + 8) + 8 * hiK);
  mn = mrow[e];

  // One scan step; no dependence on t inside -- only loop-carried pointers.
  auto step = [&](auto DO_HEAD, auto DO_PREF) {
    constexpr bool doHead = decltype(DO_HEAD)::value;
    constexpr bool doPref = decltype(DO_PREF)::value;

#pragma unroll
    for (int q = 0; q < 16; ++q) xc[q] = xn[q];
    mc = mn;
    if constexpr (doPref) {
      xrow += (size_t)N * HDIM;
      mrow += N;
#pragma unroll
      for (int q = 0; q < 16; ++q)
        xn[q] = __builtin_nontemporal_load(xrow + e * HDIM + ((q < 8) ? q : q + 8) + 8 * hiK);
      mn = mrow[e];
    }

    // ---- stage h_{t-1} (D layout) into LDS as h^T ----
#pragma unroll
    for (int r = 0; r < 8; ++r) {
      hT[(r + 8 * hiK) * ROWS + e]      = h0[r];
      hT[(16 + r + 8 * hiK) * ROWS + e] = h1[r];
    }
    mlds[e] = mc;   // lanes e and e+16 write the same value; no exec churn
    __builtin_amdgcn_wave_barrier();

    // B-layout: lane L holds B column N=L%16; element q holds K=((q<8)?q:q+8)+8*(L/16)
    float f[16], mm[16];
    {
#pragma unroll
      for (int q = 0; q < 16; ++q) {
        f[q]  = hT[(((q < 8) ? q : q + 8) + 8 * hiK) * ROWS + e];
        mm[q] = mlds[e];
      }
    }
    v16h bh, bhm, bx;
#pragma unroll
    for (int q = 0; q < 16; ++q) {
      bh[q]  = (_Float16)f[q];
      bhm[q] = (_Float16)(f[q] * mm[q]);
      bx[q]  = (_Float16)xc[q];
    }

    // ---- deferred head GEMM for step t-1 (bh holds h_{t-1}^T) ----
    if constexpr (doHead) {
      v8f hd = __builtin_amdgcn_wmma_f32_16x16x32_f16(false, ahd, false, bh,
                                                      (short)0, chd, false, false);
      v2f o; o.x = hd[0]; o.y = hd[1];
      __builtin_nontemporal_store(o, (v2f*)(orow + voff));
      orow += ostep;
    }

    // ---- preact^T = W_ih x^T + b  ;  += W_hh (h*m)^T ----
    v8f p0 = __builtin_amdgcn_wmma_f32_16x16x32_f16(false, aih0, false, bx,  (short)0, cb0, false, false);
    v8f p1 = __builtin_amdgcn_wmma_f32_16x16x32_f16(false, aih1, false, bx,  (short)0, cb1, false, false);
    p0     = __builtin_amdgcn_wmma_f32_16x16x32_f16(false, ahh0, false, bhm, (short)0, p0,  false, false);
    p1     = __builtin_amdgcn_wmma_f32_16x16x32_f16(false, ahh1, false, bhm, (short)0, p1,  false, false);

#pragma unroll
    for (int r = 0; r < 8; ++r) { h0[r] = fast_tanh(p0[r]); h1[r] = fast_tanh(p1[r]); }

    __builtin_amdgcn_wave_barrier();   // keep this iter's LDS reads before next stores
  };

  // t = 0: no head output yet (branch-free steady-state loop follows)
  step(bconst<false>{}, bconst<true>{});
  for (int t = 1; t < T - 1; ++t)
    step(bconst<true>{}, bconst<true>{});
  // t = T-1: last step, nothing left to prefetch
  step(bconst<true>{}, bconst<false>{});

  // ---- epilogue: head for t = T-1 ----
#pragma unroll
  for (int r = 0; r < 8; ++r) {
    hT[(r + 8 * hiK) * ROWS + e]      = h0[r];
    hT[(16 + r + 8 * hiK) * ROWS + e] = h1[r];
  }
  __builtin_amdgcn_wave_barrier();
  {
    v16h bh;
#pragma unroll
    for (int q = 0; q < 16; ++q)
      bh[q] = (_Float16)hT[(((q < 8) ? q : q + 8) + 8 * hiK) * ROWS + e];
    v8f hd = __builtin_amdgcn_wmma_f32_16x16x32_f16(false, ahd, false, bh,
                                                    (short)0, chd, false, false);
    v2f o; o.x = hd[0]; o.y = hd[1];
    __builtin_nontemporal_store(o, (v2f*)(orow + voff));
  }

  // ---- h_fin ----
#pragma unroll
  for (int r = 0; r < 8; ++r) {
    out_h[(size_t)(base + e) * HDIM + r + 8 * hiK]      = h0[r];
    out_h[(size_t)(base + e) * HDIM + 16 + r + 8 * hiK] = h1[r];
  }
}

extern "C" void kernel_launch(void* const* d_in, const int* in_sizes, int n_in,
                              void* d_out, int out_size, void* d_ws, size_t ws_size,
                              hipStream_t stream) {
  (void)n_in; (void)out_size; (void)d_ws; (void)ws_size;
  const float* obs   = (const float*)d_in[0];
  const float* h_obs = (const float*)d_in[1];
  const float* masks = (const float*)d_in[2];
  const float* W_ih  = (const float*)d_in[3];
  const float* W_hh  = (const float*)d_in[4];
  const float* b_ih  = (const float*)d_in[5];
  const float* b_hh  = (const float*)d_in[6];
  const float* W_mu  = (const float*)d_in[7];
  const float* b_mu  = (const float*)d_in[8];
  const float* W_lv  = (const float*)d_in[9];
  const float* b_lv  = (const float*)d_in[10];

  const int N = in_sizes[1] / HDIM;                 // 2048
  const int T = in_sizes[0] / (N * HDIM);           // 2048

  dim3 grid(N / 16), block(32);
  rnn_scan_kernel<<<grid, block, 0, stream>>>(obs, h_obs, masks, W_ih, W_hh,
                                              b_ih, b_hh, W_mu, b_mu, W_lv, b_lv,
                                              (float*)d_out, N, T);
}
